// QCNN_10849087390001
// MI455X (gfx1250) — matrix operand order repaired and checked by reading, not compiled, not run on hardware
//
#include <hip/hip_runtime.h>
#include <math.h>

// ---------------------------------------------------------------------------
// QCNN forward on gfx1250:
//   * whole 10-layer 4-qubit circuit hoisted into ONE 16x16 complex unitary
//     (kernel 1, single block), stored real-ified as a 32x32 WMMA B matrix.
//   * per-patch state init + [16 patches,32] x [32,32] GEMM via split-f16
//     emulated-f32 WMMA (6x v_wmma_f32_16x16x32_f16 per wave, kernel 2).
//   * <Z_q> via Walsh-Hadamard butterfly (ds_swizzle xor) over 16 lanes:
//     Z-expectations are WHT coefficients k = 8,4,2,1.
// ---------------------------------------------------------------------------

typedef __attribute__((ext_vector_type(16))) _Float16 v16h;
typedef __attribute__((ext_vector_type(8)))  float    v8f;

#define NLAYERS 10
#define DI 127
#define DJ 127
#define NB 16
#define NPATCH (NB * DI * DJ)      // 258064, divisible by 16
#define PI_F 3.14159265358979323846f

struct Cpx { float re, im; };
__device__ __forceinline__ Cpx cmul(Cpx a, Cpx b) {
    return { a.re * b.re - a.im * b.im, a.re * b.im + a.im * b.re };
}

// split f32 into hi/lo f16 pair: x ~= (float)h + (float)l, |l| <= ulp_f16(x)/2
__device__ __forceinline__ void split_f16(float x, _Float16& h, _Float16& l) {
    h = (_Float16)x;
    l = (_Float16)(x - (float)h);
}

// ---------------------------------------------------------------------------
// Kernel 1: build U_total = Prod_l ( D_CZ * (U_l0 (x) U_l1 (x) U_l2 (x) U_l3) )
// and emit B[K][n] = realified(U_total)[n][K]  (32x32 f32, block re/im form)
// ---------------------------------------------------------------------------
__global__ void build_unitary(const float* __restrict__ gates,   // [10][4][3]
                              const float* __restrict__ angles,  // [10][4]
                              float* __restrict__ Bmat)          // [32][32]
{
    __shared__ float Mre[256], Mim[256], Lre[256], Lim[256];
    const int t = threadIdx.x;            // 256 threads
    const int i = t >> 4, j = t & 15;     // entry (row i, col j) of 16x16

    Mre[t] = (i == j) ? 1.f : 0.f;        // M = identity
    Mim[t] = 0.f;
    __syncthreads();

    for (int l = 0; l < NLAYERS; ++l) {
        // per-qubit 2x2 unitaries U = RZ(az)*RY(ay)*RX(ax), A = onehot*angle
        Cpx U[4][2][2];
        for (int q = 0; q < 4; ++q) {
            const float ang = angles[l * 4 + q];
            const float gx = gates[(l * 4 + q) * 3 + 0];
            const float gy = gates[(l * 4 + q) * 3 + 1];
            const float gz = gates[(l * 4 + q) * 3 + 2];
            const float ax = 0.5f * gx * ang;
            const float ay = 0.5f * gy * ang;
            const float az = 0.5f * gz * ang;
            float cx, sx, cy, sy, czv, szv;
            __sincosf(ax, &sx, &cx);
            __sincosf(ay, &sy, &cy);
            __sincosf(az, &szv, &czv);
            // RY*RX
            Cpx m00 = {  cy * cx,  sy * sx };
            Cpx m01 = { -sy * cx, -cy * sx };
            Cpx m10 = {  sy * cx, -cy * sx };
            Cpx m11 = {  cy * cx, -sy * sx };
            Cpx ezm = { czv, -szv };   // e^{-i az}
            Cpx ezp = { czv,  szv };   // e^{+i az}
            U[q][0][0] = cmul(ezm, m00);
            U[q][0][1] = cmul(ezm, m01);
            U[q][1][0] = cmul(ezp, m10);
            U[q][1][1] = cmul(ezp, m11);
        }
        // Kron entry with CZ(0,1), CZ(2,3), CZ(1,2) row sign folded in.
        // state index = a*8 + b*4 + c*2 + d  (qubit 0 = most significant)
        const int ia = (i >> 3) & 1, ib = (i >> 2) & 1, ic = (i >> 1) & 1, id = i & 1;
        const int ja = (j >> 3) & 1, jb = (j >> 2) & 1, jc = (j >> 1) & 1, jd = j & 1;
        Cpx v = cmul(cmul(U[0][ia][ja], U[1][ib][jb]),
                     cmul(U[2][ic][jc], U[3][id][jd]));
        const float s = ((ia & ib) ^ (ic & id) ^ (ib & ic)) ? -1.f : 1.f;
        Lre[t] = s * v.re;
        Lim[t] = s * v.im;
        __syncthreads();
        // M = L * M
        float accr = 0.f, acci = 0.f;
        for (int k = 0; k < 16; ++k) {
            const float lr = Lre[i * 16 + k], li = Lim[i * 16 + k];
            const float mr = Mre[k * 16 + j], mi = Mim[k * 16 + j];
            accr += lr * mr - li * mi;
            acci += lr * mi + li * mr;
        }
        __syncthreads();
        Mre[t] = accr;
        Mim[t] = acci;
        __syncthreads();
    }

    // realified block form:  [yr;yi] = [[Ur,-Ui],[Ui,Ur]] [xr;xi]
    // store transposed for WMMA B:  B[K][n] = Mblock[n][K]
    const float ur = Mre[t];   // Re U[i][j]  (row n=i, col K=j)
    const float ui = Mim[t];
    const int n = i, K = j;
    Bmat[K * 32 + n]                =  ur;
    Bmat[(K + 16) * 32 + n]         = -ui;
    Bmat[K * 32 + (n + 16)]         =  ui;
    Bmat[(K + 16) * 32 + (n + 16)]  =  ur;
}

// xor-butterfly stage over 32 lanes via ds_swizzle (offset = xor<<10 | 0x1f)
#define WHT_STAGE(MASK, OFFS)                                                \
    {                                                                        \
        _Pragma("unroll")                                                    \
        for (int r = 0; r < 8; ++r) {                                        \
            const int   iv = __float_as_int(pr[r]);                          \
            const float o  = __int_as_float(                                 \
                                 __builtin_amdgcn_ds_swizzle(iv, (OFFS)));   \
            pr[r] = (col & (MASK)) ? (o - pr[r]) : (pr[r] + o);              \
        }                                                                    \
    }

// ---------------------------------------------------------------------------
// Kernel 2: 16 patches per wave. A = init states (16x32, block re/im),
// split-f16 WMMA GEMM gives y_real / y_imag tiles at ~f32 accuracy,
// probs = per-lane sum of squares, <Z_q> = WHT coefficients 8,4,2,1.
// ---------------------------------------------------------------------------
__global__ void __launch_bounds__(256)
qcnn_forward(const float* __restrict__ images,   // [16][128][128][3]
             const float* __restrict__ Bmat,     // [32][32]
             float* __restrict__ out)            // [NPATCH][4]
{
    const int lane  = threadIdx.x & 31;
    const int wave  = threadIdx.x >> 5;
    const int gwave = blockIdx.x * 8 + wave;     // 256 threads = 8 waves
    const int base  = gwave * 16;                // first patch of wave tile

    const int hi  = lane >> 4;                   // 0: lanes 0-15, 1: 16-31
    const int col = lane & 15;

    // ---- B fragments, split into hi/lo f16 (K = 8*hi + e for e<8, 16 + 8*hi + (e-8) for e>=8; two tiles) ----
    v16h bh0, bl0, bh1, bl1;
    #pragma unroll
    for (int e = 0; e < 16; ++e) {
        const int K = (e < 8) ? (8 * hi + e) : (16 + 8 * hi + (e - 8));
        _Float16 h, l;
        split_f16(Bmat[K * 32 + col], h, l);         // y_real columns 0..15
        bh0[e] = h; bl0[e] = l;
        split_f16(Bmat[K * 32 + col + 16], h, l);    // y_imag columns 16..31
        bh1[e] = h; bl1[e] = l;
    }

    // ---- init state for this lane's patch (A fragment) ----
    int p = base + col;
    if (p >= NPATCH) p = NPATCH - 1;             // clamp loads, keep EXEC full
    const int bi  = p / (DI * DJ);
    const int rem = p - bi * (DI * DJ);
    const int ii  = rem / DJ;
    const int jj  = rem - ii * DJ;

    Cpx S[4][2];
    #pragma unroll
    for (int a2 = 0; a2 < 2; ++a2)
    #pragma unroll
    for (int b2 = 0; b2 < 2; ++b2) {
        const int q = a2 * 2 + b2;
        const float* px = images + (((bi * 128) + (ii + a2)) * 128 + (jj + b2)) * 3;
        const float phi = PI_F * px[0];
        const float th  = PI_F * px[1];
        const float om  = PI_F * px[2];
        float ct, st2, cpo, spo, cmo, smo;
        __sincosf(0.5f * th,         &st2, &ct);
        __sincosf(0.5f * (phi + om), &spo, &cpo);
        __sincosf(0.5f * (om - phi), &smo, &cmo);
        S[q][0] = { cpo * ct,  -spo * ct  };     // amplitude on |0>
        S[q][1] = { cmo * st2,  smo * st2 };     // amplitude on |1>
    }

    // lane's 8 amplitudes: j = hi*8 + (b*4 + c*2 + d)
    Cpx amp[8];
    {
        const Cpx t0 = S[0][hi];
        const Cpx t01[2] = { cmul(t0, S[1][0]), cmul(t0, S[1][1]) };
        #pragma unroll
        for (int b2 = 0; b2 < 2; ++b2)
        #pragma unroll
        for (int c2 = 0; c2 < 2; ++c2) {
            const Cpx tt = cmul(t01[b2], S[2][c2]);
            amp[b2 * 4 + c2 * 2 + 0] = cmul(tt, S[3][0]);
            amp[b2 * 4 + c2 * 2 + 1] = cmul(tt, S[3][1]);
        }
    }

    // A fragment (split hi/lo): elements 0-7 = reals, 8-15 = imags
    v16h ah, al;
    #pragma unroll
    for (int e = 0; e < 8; ++e) {
        _Float16 h, l;
        split_f16(amp[e].re, h, l);  ah[e]     = h;  al[e]     = l;
        split_f16(amp[e].im, h, l);  ah[e + 8] = h;  al[e + 8] = l;
    }

    // ---- D = (Ah+Al) x (Bh+Bl) ~ Ah*Bl + Al*Bh + Ah*Bh (f32 accumulate) ----
    v8f d0v = {}, d1v = {};
    d0v = __builtin_amdgcn_wmma_f32_16x16x32_f16(false, ah, false, bl0,
                                                 (short)0, d0v, false, false);
    d0v = __builtin_amdgcn_wmma_f32_16x16x32_f16(false, al, false, bh0,
                                                 (short)0, d0v, false, false);
    d0v = __builtin_amdgcn_wmma_f32_16x16x32_f16(false, ah, false, bh0,
                                                 (short)0, d0v, false, false);
    d1v = __builtin_amdgcn_wmma_f32_16x16x32_f16(false, ah, false, bl1,
                                                 (short)0, d1v, false, false);
    d1v = __builtin_amdgcn_wmma_f32_16x16x32_f16(false, al, false, bh1,
                                                 (short)0, d1v, false, false);
    d1v = __builtin_amdgcn_wmma_f32_16x16x32_f16(false, ah, false, bh1,
                                                 (short)0, d1v, false, false);

    // probs: lane holds |y[col]|^2 for patches r + 8*hi (VGPR r = patch row)
    v8f pr;
    #pragma unroll
    for (int r = 0; r < 8; ++r)
        pr[r] = d0v[r] * d0v[r] + d1v[r] * d1v[r];

    // Walsh-Hadamard butterfly over the 16-lane half (j = col):
    // afterwards lane k holds sum_j (-1)^{popcount(j&k)} p_j
    WHT_STAGE(1, 0x041F)   // xor 1
    WHT_STAGE(2, 0x081F)   // xor 2
    WHT_STAGE(4, 0x101F)   // xor 4
    WHT_STAGE(8, 0x201F)   // xor 8

    // <Z_0> = coeff k=8 (sign by qubit-0 bit), <Z_1>=k=4, <Z_2>=k=2, <Z_3>=k=1
    int q = -1;
    if      (col == 8) q = 0;
    else if (col == 4) q = 1;
    else if (col == 2) q = 2;
    else if (col == 1) q = 3;
    if (q >= 0) {
        #pragma unroll
        for (int r = 0; r < 8; ++r) {
            const int pp = base + r + 8 * hi;
            if (pp < NPATCH) out[pp * 4 + q] = pr[r];
        }
    }
}

// ---------------------------------------------------------------------------
extern "C" void kernel_launch(void* const* d_in, const int* in_sizes, int n_in,
                              void* d_out, int out_size, void* d_ws, size_t ws_size,
                              hipStream_t stream) {
    const float* images = (const float*)d_in[0];   // [16,128,128,3] f32
    const float* gates  = (const float*)d_in[1];   // [10,4,3] f32
    const float* angles = (const float*)d_in[2];   // [10,4] f32
    float* out  = (float*)d_out;                   // [16,127,127,4] f32
    float* Bmat = (float*)d_ws;                    // 32*32 f32 = 4KB scratch

    build_unitary<<<1, 256, 0, stream>>>(gates, angles, Bmat);

    const int waves  = (NPATCH + 15) / 16;         // 16129
    const int blocks = (waves + 7) / 8;            // 2017 (256-thread blocks)
    qcnn_forward<<<blocks, 256, 0, stream>>>(images, Bmat, out);
}
